// Head_23098334118438
// MI455X (gfx1250) — matrix operand order repaired and checked by reading, not compiled, not run on hardware
//
#include <hip/hip_runtime.h>

typedef _Float16 h16;
typedef __attribute__((ext_vector_type(16))) _Float16 v16h;
typedef __attribute__((ext_vector_type(8)))  _Float16 v8h;
typedef __attribute__((ext_vector_type(8)))  float    v8f;
typedef __attribute__((ext_vector_type(4)))  float    v4f;

#define NB 4
#define NT 4096
#define ND 1024
#define NH 64

union V16 { v16h v; v8h h[2]; };
union H8U { _Float16 h[8]; uint4 u; };

// ---------------------------------------------------------------------------
// Kernel A: convert x (f32) -> xh (f16), one pass. 8 elems / thread.
// ---------------------------------------------------------------------------
__global__ void __launch_bounds__(256)
xcvt_kernel(const float* __restrict__ x, h16* __restrict__ xh) {
  size_t i = ((size_t)blockIdx.x * 256 + threadIdx.x) * 8;
  v4f x0 = *(const v4f*)(x + i);
  v4f x1 = *(const v4f*)(x + i + 4);
  H8U pk;
#pragma unroll
  for (int j = 0; j < 4; ++j) {
    pk.h[j]     = (h16)x0[j];
    pk.h[4 + j] = (h16)x1[j];
  }
  *(uint4*)(xh + i) = pk.u;
}

// ---------------------------------------------------------------------------
// Kernel B: convert + transpose weights: Wt[m][h][d] = (f16) W_m[d][h]
// ---------------------------------------------------------------------------
__global__ void __launch_bounds__(256)
wt_kernel(const float* __restrict__ Wk, const float* __restrict__ Wq,
          const float* __restrict__ Wv, h16* __restrict__ wt) {
  int idx = blockIdx.x * 256 + threadIdx.x;          // 3*D*H threads exactly
  int m   = idx / (ND * NH);
  int rem = idx % (ND * NH);
  int d   = rem / NH;
  int h   = rem % NH;
  const float* W = (m == 0) ? Wk : (m == 1 ? Wq : Wv);
  wt[(size_t)m * NH * ND + (size_t)h * ND + d] = (h16)W[(size_t)d * NH + h];
}

// ---------------------------------------------------------------------------
// Kernel C: projections via WMMA, software-pipelined.
// One wave per 16-row tile computes all 12 output tiles (3 matrices x 4 cols),
// A double-buffered, B rotated through 4 buffers loaded 4 tiles ahead.
// ---------------------------------------------------------------------------
__global__ void __launch_bounds__(256)
proj_kernel(const h16* __restrict__ xh, const h16* __restrict__ wt,
            h16* __restrict__ qh, h16* __restrict__ kh, h16* __restrict__ vT) {
  int wid  = blockIdx.x * 8 + (threadIdx.x >> 5);   // rt in [0, 4096)
  int lane = threadIdx.x & 31;
  int lo = lane & 15, hi = lane >> 4;

  int rt = wid;
  const h16* xrow = xh + (size_t)(rt * 16 + lo) * ND;

  // per-tile weight column base: mat = t>>2, ct = t&3 (constant tile strides)
  const h16* wbase[12];
#pragma unroll
  for (int t = 0; t < 12; ++t) {
    int mat = t >> 2, ct = t & 3;
    wbase[t] = wt + (size_t)mat * NH * ND + (size_t)(ct * 16 + lo) * ND + 16 * hi;
  }

  v8f acc[12] = {};
  v16h b[4];
  V16 a0, a1;

  a0.h[0] = *(const v8h*)(xrow + 16 * hi);
  a0.h[1] = *(const v8h*)(xrow + 16 * hi + 8);
#pragma unroll
  for (int t = 0; t < 4; ++t) b[t] = *(const v16h*)(wbase[t]);

  // one dk-group: 12 WMMAs, refill B buffers 4 tiles ahead
  auto group = [&](const V16& av, int dkcur, bool refill_next) {
#pragma unroll
    for (int t = 0; t < 12; ++t) {
      acc[t] = __builtin_amdgcn_wmma_f32_16x16x32_f16(false, av.v, false, b[t & 3],
                                                      (short)0, acc[t], false, false);
      if (t + 4 < 12)
        b[t & 3] = *(const v16h*)(wbase[t + 4] + dkcur);
      else if (refill_next)
        b[t & 3] = *(const v16h*)(wbase[t - 8] + dkcur + 32);
    }
  };

  int dk = 0;
  for (; dk < ND - 64; dk += 64) {
    a1.h[0] = *(const v8h*)(xrow + dk + 32 + 16 * hi);
    a1.h[1] = *(const v8h*)(xrow + dk + 32 + 16 * hi + 8);
    group(a0, dk, true);
    a0.h[0] = *(const v8h*)(xrow + dk + 64 + 16 * hi);
    a0.h[1] = *(const v8h*)(xrow + dk + 64 + 16 * hi + 8);
    group(a1, dk + 32, true);
  }
  // tail: dk == ND-64
  a1.h[0] = *(const v8h*)(xrow + dk + 32 + 16 * hi);
  a1.h[1] = *(const v8h*)(xrow + dk + 32 + 16 * hi + 8);
  group(a0, dk, true);
  group(a1, dk + 32, false);

  int bb = (rt * 16) / NT;
  int t0 = (rt * 16) % NT + 8 * hi;
  int rbase = rt * 16 + 8 * hi;
#pragma unroll
  for (int t = 0; t < 12; ++t) {
    int mat = t >> 2, ct = t & 3;
    int col = ct * 16 + lo;
    if (mat == 2) {
      // V transposed: vT[b][h][t]; C rows r=0..7 contiguous in t
      H8U pk;
#pragma unroll
      for (int r = 0; r < 8; ++r) pk.h[r] = (h16)acc[t][r];
      *(uint4*)(vT + ((size_t)(bb * NH + col)) * NT + t0) = pk.u;
    } else {
      h16* dst = (mat == 1) ? qh : kh;
      float sc = (mat == 1) ? 0.125f : 1.0f;   // fold H^-0.5 into Q
#pragma unroll
      for (int r = 0; r < 8; ++r)
        dst[(size_t)(rbase + r) * NH + col] = (h16)(acc[t][r] * sc);
    }
  }
}

// ---------------------------------------------------------------------------
// Kernel D: causal flash attention. One 16-query tile per wave, 32-key chunks.
// All K/V B-tiles issued at chunk start so V waits hide under softmax VALU.
// ---------------------------------------------------------------------------
__global__ void __launch_bounds__(256)
attn_kernel(const h16* __restrict__ qh, const h16* __restrict__ kh,
            const h16* __restrict__ vT, float* __restrict__ out) {
  __shared__ __align__(16) _Float16 pbuf[8][16 * 40];   // pitch 40 f16 = 80B
  int w    = threadIdx.x >> 5;
  int lane = threadIdx.x & 31;
  int lo = lane & 15, hi = lane >> 4;

  int task  = blockIdx.x * 8 + w;     // B * T/16 = 1024 tasks
  int b     = task >> 8;
  int qt    = task & 255;
  int qbase = qt * 16;

  const h16* qb = qh + ((size_t)(b * NT + qbase)) * NH;
  const h16* kb = kh + ((size_t)b * NT) * NH;
  const h16* vb = vT + ((size_t)b * NH) * NT;

  // Q tile in A-layout, both H-halves (H=64 -> 2 WMMA contraction chunks)
  V16 aq[2];
  {
    const h16* qrow = qb + (size_t)lo * NH;
#pragma unroll
    for (int hs = 0; hs < 2; ++hs) {
      aq[hs].h[0] = *(const v8h*)(qrow + hs * 32 + 16 * hi);
      aq[hs].h[1] = *(const v8h*)(qrow + hs * 32 + 16 * hi + 8);
    }
  }

  float mrow[8], lrow[8];
  v8f o[4] = {};
#pragma unroll
  for (int r = 0; r < 8; ++r) { mrow[r] = -3.0e38f; lrow[r] = 0.0f; }

  const int qlast = qbase + 15;
  for (int c = 0; c <= qlast; c += 32) {
    if (c + 32 <= qlast) {
      __builtin_prefetch(kb + (size_t)(c + 32 + lane) * NH, 0, 1);
      __builtin_prefetch(vb + (size_t)lane * NT + c + 32, 0, 1);
    }
    // ---- issue ALL K and V tile loads for this chunk up front ----
    v16h bk[4], bv[4];
    {
      const h16* kr0 = kb + (size_t)(c + lo) * NH;
      const h16* kr1 = kb + (size_t)(c + 16 + lo) * NH;
      bk[0] = *(const v16h*)(kr0 + 16 * hi);        // t=0, H 0..31
      bk[1] = *(const v16h*)(kr0 + 32 + 16 * hi);   // t=0, H 32..63
      bk[2] = *(const v16h*)(kr1 + 16 * hi);        // t=1, H 0..31
      bk[3] = *(const v16h*)(kr1 + 32 + 16 * hi);   // t=1, H 32..63
#pragma unroll
      for (int nb = 0; nb < 4; ++nb)
        bv[nb] = *(const v16h*)(vb + (size_t)(nb * 16 + lo) * NT + c + 16 * hi);
    }
    // ---- S = (Q*scale) . K^T : two 16-key tiles ----
    v8f s[2];
#pragma unroll
    for (int t = 0; t < 2; ++t) {
      v8f st = {};
      st = __builtin_amdgcn_wmma_f32_16x16x32_f16(false, aq[0].v, false, bk[2 * t],
                                                  (short)0, st, false, false);
      st = __builtin_amdgcn_wmma_f32_16x16x32_f16(false, aq[1].v, false, bk[2 * t + 1],
                                                  (short)0, st, false, false);
      s[t] = st;
    }
    // ---- causal mask + row max (rows live in half-waves, width-16 xor) ----
    float mx[8];
#pragma unroll
    for (int r = 0; r < 8; ++r) {
      int rowg = qbase + 8 * hi + r;
#pragma unroll
      for (int t = 0; t < 2; ++t) {
        int colg = c + t * 16 + lo;
        s[t][r] = (colg <= rowg) ? s[t][r] : -3.0e38f;
      }
      mx[r] = fmaxf(s[0][r], s[1][r]);
    }
#pragma unroll
    for (int off = 1; off < 16; off <<= 1)
#pragma unroll
      for (int r = 0; r < 8; ++r)
        mx[r] = fmaxf(mx[r], __shfl_xor(mx[r], off, 16));
    // ---- online softmax update ----
    float alpha[8], rs[8], p0[8], p1[8];
#pragma unroll
    for (int r = 0; r < 8; ++r) {
      float mn = fmaxf(mrow[r], mx[r]);
      alpha[r] = __expf(mrow[r] - mn);
      mrow[r]  = mn;
      p0[r] = __expf(s[0][r] - mn);
      p1[r] = __expf(s[1][r] - mn);
      rs[r] = p0[r] + p1[r];
    }
#pragma unroll
    for (int off = 1; off < 16; off <<= 1)
#pragma unroll
      for (int r = 0; r < 8; ++r)
        rs[r] += __shfl_xor(rs[r], off, 16);
#pragma unroll
    for (int r = 0; r < 8; ++r) {
      lrow[r] = lrow[r] * alpha[r] + rs[r];
#pragma unroll
      for (int nb = 0; nb < 4; ++nb) o[nb][r] *= alpha[r];
    }
    // ---- C-layout P -> LDS (row-major) -> A-layout reload ----
#pragma unroll
    for (int r = 0; r < 8; ++r) {
      int rl = 8 * hi + r;
      pbuf[w][rl * 40 + lo]      = (h16)p0[r];
      pbuf[w][rl * 40 + 16 + lo] = (h16)p1[r];
    }
    V16 ap;
    ap.h[0] = *(const v8h*)&pbuf[w][lo * 40 + 16 * hi];
    ap.h[1] = *(const v8h*)&pbuf[w][lo * 40 + 16 * hi + 8];
    // ---- O += P . V  (V^T gives contraction-contiguous B columns) ----
#pragma unroll
    for (int nb = 0; nb < 4; ++nb)
      o[nb] = __builtin_amdgcn_wmma_f32_16x16x32_f16(false, ap.v, false, bv[nb],
                                                     (short)0, o[nb], false, false);
  }
  // ---- epilogue: O / l ----
#pragma unroll
  for (int r = 0; r < 8; ++r) {
    float inv = 1.0f / lrow[r];
    int rowg = qbase + 8 * hi + r;
    float* orow = out + ((size_t)(b * NT + rowg)) * NH;
#pragma unroll
    for (int nb = 0; nb < 4; ++nb)
      orow[nb * 16 + lo] = o[nb][r] * inv;
  }
}

// ---------------------------------------------------------------------------
extern "C" void kernel_launch(void* const* d_in, const int* in_sizes, int n_in,
                              void* d_out, int out_size, void* d_ws, size_t ws_size,
                              hipStream_t stream) {
  const float* x  = (const float*)d_in[0];
  const float* Wk = (const float*)d_in[1];
  const float* Wq = (const float*)d_in[2];
  const float* Wv = (const float*)d_in[3];
  float* out = (float*)d_out;

  char* ws = (char*)d_ws;
  const size_t nQ = (size_t)NB * NT * NH;          // 1M elems
  const size_t nX = (size_t)NB * NT * ND;          // 16M elems
  h16* qh = (h16*)(ws);                            // 2 MB
  h16* kh = (h16*)(ws + nQ * 2);                   // 2 MB
  h16* vT = (h16*)(ws + nQ * 4);                   // 2 MB, layout [b][h][t]
  h16* wt = (h16*)(ws + nQ * 6);                   // 384 KB, [m][h][d]
  h16* xh = (h16*)(ws + nQ * 6 + (size_t)3 * NH * ND * 2);  // 32 MB

  xcvt_kernel<<<(int)(nX / 8 / 256), 256, 0, stream>>>(x, xh);
  wt_kernel  <<<(3 * ND * NH) / 256,  256, 0, stream>>>(Wk, Wq, Wv, wt);
  proj_kernel<<<4096 / 8,             256, 0, stream>>>(xh, wt, qh, kh, vT);
  attn_kernel<<<(NB * (NT / 16)) / 8, 256, 0, stream>>>(qh, kh, vT, out);
}
